// Three_Layer_Model_40123584479552
// MI455X (gfx1250) — matrix operand order repaired and checked by reading, not compiled, not run on hardware
//
#include <hip/hip_runtime.h>

// Problem constants (from the reference)
#define BB    2048
#define ISZ   512
#define NN    2048
#define OSZ   128

typedef __attribute__((ext_vector_type(16))) __bf16        v16bf;
typedef __attribute__((ext_vector_type(8)))  float         v8f;
typedef __attribute__((ext_vector_type(4)))  unsigned int  u32x4;

__device__ __forceinline__ unsigned short f2bf(float f) {
  // round-to-nearest-even f32 -> bf16
  unsigned int u = __float_as_uint(f);
  unsigned int r = u + 0x7FFFu + ((u >> 16) & 1u);
  return (unsigned short)(r >> 16);
}

// ---------------------------------------------------------------- prologue ---
__global__ void cvt_bf16_kernel(const float* __restrict__ src,
                                unsigned short* __restrict__ dst, int n) {
  int i = blockIdx.x * blockDim.x + threadIdx.x;
  if (i < n) dst[i] = f2bf(src[i]);
}

// W: R x C f32 row-major.  Writes Wt: C x R bf16 row-major, with optional
// elementwise mask and/or zeroed diagonal (applied in the ORIGINAL layout).
__global__ void transpose_mask_bf16_kernel(const float* __restrict__ W,
                                           const float* __restrict__ mask,
                                           int zero_diag, int R, int C,
                                           unsigned short* __restrict__ Wt) {
  __shared__ float tile[32][33];
  const int c0 = blockIdx.x * 32, r0 = blockIdx.y * 32;
  const int tx = threadIdx.x, ty = threadIdx.y;
#pragma unroll
  for (int i = 0; i < 4; ++i) {
    int r = r0 + ty + i * 8, c = c0 + tx;
    float v = W[(size_t)r * C + c];
    if (mask) v *= mask[(size_t)r * C + c];
    if (zero_diag && r == c) v = 0.0f;
    tile[ty + i * 8][tx] = v;
  }
  __syncthreads();
#pragma unroll
  for (int i = 0; i < 4; ++i) {
    int c = c0 + ty + i * 8;  // output row = original column
    int r = r0 + tx;          // output col = original row
    Wt[(size_t)c * R + r] = f2bf(tile[tx][ty + i * 8]);
  }
}

// ----------------------------------------------------------- fused bf16 GEMM -
// C(128x128 per block) = A1[M,K1] @ B1t^T + A2[M,K2] @ B2t^T, with B*t stored
// transposed as [N][K] bf16.  Epilogue (FUSED):
//   out = h_in*0.9 + relu(C + bias)*0.1   (f32 to outF, bf16 to outB)
// else plain f32 store.
#define LDK 40  // padded LDS row stride (shorts) for a 32-wide K tile

// One async 16B global->LDS move per lane (CDNA5 GLOBAL_LOAD_ASYNC_TO_LDS,
// tracked by ASYNCcnt; no VGPR round-trip, overlaps with WMMA compute).
__device__ __forceinline__ void async_cp16(const unsigned short* gsrc,
                                           unsigned short* ldst) {
  unsigned lo = (unsigned)(size_t)ldst;  // flat low 32 bits == LDS byte addr
  asm volatile("global_load_async_to_lds_b128 %0, %1, off"
               :: "v"(lo), "v"(gsrc) : "memory");
}
__device__ __forceinline__ void wait_async0() {
  asm volatile("s_wait_asynccnt 0x0" ::: "memory");
}

// Stage one 128x32 A tile + 128x32 B tile into LDS buffer `buf` (async).
__device__ __forceinline__ void stage_tile(const unsigned short* __restrict__ A,
                                           const unsigned short* __restrict__ Bt,
                                           int K, int kt, int m0, int n0, int t,
                                           unsigned short* As,
                                           unsigned short* Bs) {
#pragma unroll
  for (int i = 0; i < 2; ++i) {
    int c   = t + i * 256;
    int row = c >> 2;       // 0..127
    int k8  = (c & 3) * 8;  // 0,8,16,24
    async_cp16(A  + (size_t)(m0 + row) * K + kt + k8, As + row * LDK + k8);
    async_cp16(Bt + (size_t)(n0 + row) * K + kt + k8, Bs + row * LDK + k8);
  }
}

template <bool FUSED>
__global__ __launch_bounds__(256, 2)
void fused_gemm_kernel(const unsigned short* __restrict__ A1, int K1,
                       const unsigned short* __restrict__ B1t,
                       const unsigned short* __restrict__ A2, int K2,
                       const unsigned short* __restrict__ B2t,
                       const float* __restrict__ h_in,
                       const float* __restrict__ bias,
                       float* __restrict__ outF,
                       unsigned short* __restrict__ outB,
                       int Ntot) {
  __shared__ __align__(16) unsigned short As[2][128 * LDK];
  __shared__ __align__(16) unsigned short Bs[2][128 * LDK];

  const int t    = threadIdx.x;
  const int lane = t & 31;
  const int w    = t >> 5;   // 8 waves
  const int wm   = w >> 1;   // 0..3  (M direction)
  const int wn   = w & 1;    // 0..1  (N direction)
  const int m0   = blockIdx.y * 128;
  const int n0   = blockIdx.x * 128;
  const int l15  = lane & 15;
  const int hi   = lane >> 4;

  v8f acc[2][4];
#pragma unroll
  for (int mi = 0; mi < 2; ++mi)
#pragma unroll
    for (int ni = 0; ni < 4; ++ni)
#pragma unroll
      for (int j = 0; j < 8; ++j) acc[mi][ni][j] = 0.0f;

  const int t1 = K1 >> 5;
  const int t2 = K2 >> 5;
  const int ttot = t1 + t2;

  // issue tile `ti` into buffer `buf`
  auto issue = [&](int ti, int buf) {
    if (ti < t1)
      stage_tile(A1, B1t, K1, ti * 32, m0, n0, t, As[buf], Bs[buf]);
    else
      stage_tile(A2, B2t, K2, (ti - t1) * 32, m0, n0, t, As[buf], Bs[buf]);
  };

  issue(0, 0);
  for (int ti = 0; ti < ttot; ++ti) {
    const int buf = ti & 1;
    wait_async0();        // this wave's copies for tile `ti` are in LDS
    __syncthreads();      // ...and everyone else's; also closes reads of buf ^ 1
    if (ti + 1 < ttot) issue(ti + 1, buf ^ 1);  // overlap copy with compute

    // build WMMA fragments per ISA bf16 layout
    const unsigned short* Ab = As[buf];
    const unsigned short* Bb = Bs[buf];
    v16bf af[2], bfr[4];
#pragma unroll
    for (int mi = 0; mi < 2; ++mi) {
      int row = wm * 32 + mi * 16 + l15;
      const u32x4* p = (const u32x4*)(Ab + row * LDK + hi * 8);
      union { u32x4 q[2]; v16bf v; } u;
      u.q[0] = p[0];  // K = hi*8 .. hi*8+7
      u.q[1] = p[2];  // K = 16+hi*8 .. 16+hi*8+7
      af[mi] = u.v;
    }
#pragma unroll
    for (int ni = 0; ni < 4; ++ni) {
      int nrow = wn * 64 + ni * 16 + l15;
      const u32x4* p = (const u32x4*)(Bb + nrow * LDK + hi * 8);
      union { u32x4 q[2]; v16bf v; } u;
      u.q[0] = p[0];  // K = hi*8 .. hi*8+7
      u.q[1] = p[2];  // K = 16+hi*8 .. 16+hi*8+7
      bfr[ni] = u.v;
    }

#pragma unroll
    for (int mi = 0; mi < 2; ++mi)
#pragma unroll
      for (int ni = 0; ni < 4; ++ni)
        acc[mi][ni] = __builtin_amdgcn_wmma_f32_16x16x32_bf16(
            false, af[mi], false, bfr[ni], (short)0, acc[mi][ni],
            false, false);
  }

  // ------------------------------------------------------------- epilogue ---
  const int colBase = n0 + wn * 64;
  const int rowBase = m0 + wm * 32;
#pragma unroll
  for (int mi = 0; mi < 2; ++mi) {
#pragma unroll
    for (int ni = 0; ni < 4; ++ni) {
      const int col  = colBase + ni * 16 + l15;
      const int row0 = rowBase + mi * 16 + hi * 8;
      if constexpr (FUSED) {
        const float b = bias[col];
        float hv[8];
#pragma unroll
        for (int r = 0; r < 8; ++r)
          hv[r] = h_in[(size_t)(row0 + r) * Ntot + col];
#pragma unroll
        for (int r = 0; r < 8; ++r) {
          size_t idx = (size_t)(row0 + r) * Ntot + col;
          float pre = acc[mi][ni][r] + b;
          pre = pre > 0.0f ? pre : 0.0f;
          float v = hv[r] * 0.9f + pre * 0.1f;  // (1-tc), tc = 0.1
          outF[idx] = v;
          outB[idx] = f2bf(v);
        }
      } else {
#pragma unroll
        for (int r = 0; r < 8; ++r)
          outF[(size_t)(row0 + r) * Ntot + col] = acc[mi][ni][r];
      }
    }
  }
}

// ------------------------------------------------------------------ launch ---
extern "C" void kernel_launch(void* const* d_in, const int* in_sizes, int n_in,
                              void* d_out, int out_size, void* d_ws,
                              size_t ws_size, hipStream_t stream) {
  (void)in_sizes; (void)n_in; (void)out_size; (void)ws_size;
  const float* x        = (const float*)d_in[0];
  const float* h0_in    = (const float*)d_in[1];
  const float* h1_in    = (const float*)d_in[2];
  const float* h2_in    = (const float*)d_in[3];
  const float* i_to_h0  = (const float*)d_in[4];
  const float* h0_w     = (const float*)d_in[5];
  const float* h0_b     = (const float*)d_in[6];
  const float* h0_to_h1 = (const float*)d_in[7];
  const float* h1_w     = (const float*)d_in[8];
  const float* h1_b     = (const float*)d_in[9];
  const float* h1_to_h2 = (const float*)d_in[10];
  const float* h2_w     = (const float*)d_in[11];
  const float* h2_b     = (const float*)d_in[12];
  const float* w_out    = (const float*)d_in[13];
  const float* m01      = (const float*)d_in[14];
  const float* m12      = (const float*)d_in[15];

  float* outF = (float*)d_out;
  const size_t BN = (size_t)BB * NN;

  unsigned short* ws = (unsigned short*)d_ws;
  size_t off = 0;
  auto take = [&](size_t n) { unsigned short* p = ws + off; off += n; return p; };
  unsigned short* Xb    = take((size_t)BB * ISZ);
  unsigned short* h0inb = take(BN);
  unsigned short* h1inb = take(BN);
  unsigned short* h2inb = take(BN);
  unsigned short* Wxt   = take((size_t)ISZ * NN);   // [N][K=512]
  unsigned short* W0t   = take((size_t)NN * NN);
  unsigned short* W01t  = take((size_t)NN * NN);
  unsigned short* W1t   = take((size_t)NN * NN);
  unsigned short* W12t  = take((size_t)NN * NN);
  unsigned short* W2t   = take((size_t)NN * NN);
  unsigned short* Woutt = take((size_t)OSZ * NN);   // [128][K=2048]
  unsigned short* h0b   = take(BN);
  unsigned short* h1b   = take(BN);
  unsigned short* h2b   = take(BN);

  // activations f32 -> bf16
  cvt_bf16_kernel<<<(BB * ISZ) / 256, 256, 0, stream>>>(x, Xb, BB * ISZ);
  cvt_bf16_kernel<<<(int)(BN / 256), 256, 0, stream>>>(h0_in, h0inb, (int)BN);
  cvt_bf16_kernel<<<(int)(BN / 256), 256, 0, stream>>>(h1_in, h1inb, (int)BN);
  cvt_bf16_kernel<<<(int)(BN / 256), 256, 0, stream>>>(h2_in, h2inb, (int)BN);

  // weights: mask/diag + transpose + bf16
  dim3 tb(32, 8);
  transpose_mask_bf16_kernel<<<dim3(NN / 32, ISZ / 32), tb, 0, stream>>>(
      i_to_h0, nullptr, 0, ISZ, NN, Wxt);
  transpose_mask_bf16_kernel<<<dim3(NN / 32, NN / 32), tb, 0, stream>>>(
      h0_w, nullptr, 1, NN, NN, W0t);
  transpose_mask_bf16_kernel<<<dim3(NN / 32, NN / 32), tb, 0, stream>>>(
      h0_to_h1, m01, 0, NN, NN, W01t);
  transpose_mask_bf16_kernel<<<dim3(NN / 32, NN / 32), tb, 0, stream>>>(
      h1_w, nullptr, 1, NN, NN, W1t);
  transpose_mask_bf16_kernel<<<dim3(NN / 32, NN / 32), tb, 0, stream>>>(
      h1_to_h2, m12, 0, NN, NN, W12t);
  transpose_mask_bf16_kernel<<<dim3(NN / 32, NN / 32), tb, 0, stream>>>(
      h2_w, nullptr, 1, NN, NN, W2t);
  transpose_mask_bf16_kernel<<<dim3(OSZ / 32, NN / 32), tb, 0, stream>>>(
      w_out, nullptr, 0, NN, OSZ, Woutt);

  // four dependent GEMM stages
  dim3 g(NN / 128, BB / 128);   // (16,16)
  fused_gemm_kernel<true><<<g, 256, 0, stream>>>(
      Xb, ISZ, Wxt, h0inb, NN, W0t, h0_in, h0_b, outF, h0b, NN);
  fused_gemm_kernel<true><<<g, 256, 0, stream>>>(
      h0b, NN, W01t, h1inb, NN, W1t, h1_in, h1_b, outF + BN, h1b, NN);
  fused_gemm_kernel<true><<<g, 256, 0, stream>>>(
      h1b, NN, W12t, h2inb, NN, W2t, h2_in, h2_b, outF + 2 * BN, h2b, NN);
  dim3 go(OSZ / 128, BB / 128); // (1,16)
  fused_gemm_kernel<false><<<go, 256, 0, stream>>>(
      h2b, NN, Woutt, nullptr, 0, nullptr, nullptr, nullptr,
      outF + 3 * BN, nullptr, OSZ);
}
